// PET_Layer_44564580663873
// MI455X (gfx1250) — compile-verified, matrix operand order checked by reading, not run on hardware
//
#include <hip/hip_runtime.h>

#define N_NODES 10000
#define N_EDGES 160000
#define DDIM    256
#define NLAYERS 2

typedef __bf16 bf16_t;
typedef bf16_t bf16x16 __attribute__((ext_vector_type(16)));
typedef float  v8f     __attribute__((ext_vector_type(8)));
typedef unsigned int u32x4 __attribute__((ext_vector_type(4)));
typedef int          i32x4 __attribute__((ext_vector_type(4)));
typedef int          i32x8 __attribute__((ext_vector_type(8)));

union FragU { uint4 u[2]; bf16x16 v; };

__device__ __forceinline__ unsigned int f32_to_bf16(float x) {
    unsigned int b = __float_as_uint(x);
    b += 0x7FFFu + ((b >> 16) & 1u);       // round-to-nearest-even
    return b >> 16;
}

// monotone float<->uint mapping for atomicMax on floats
__device__ __forceinline__ unsigned int enc_f32(float x) {
    unsigned int u = __float_as_uint(x);
    return ((int)u < 0) ? ~u : (u | 0x80000000u);
}
__device__ __forceinline__ float dec_f32(unsigned int u) {
    return __uint_as_float((u & 0x80000000u) ? (u & 0x7FFFFFFFu) : ~u);
}

// ---------------------------------------------------------------------------
// TDM: 2D tile load Global->LDS. Elements are 2 bytes (bf16).
// Tile rows are contiguous in LDS: LDS layout = [tile_h][tile_w] bf16.
// D# packing per CDNA5 ISA 8.3/8.4 (group0 128b, group1 256b), groups 2/3 zero.
// ---------------------------------------------------------------------------
__device__ __forceinline__ void tdm_load_2d_bf16(unsigned lds_addr, const void* gaddr,
                                                 unsigned tile_w, unsigned tile_h,
                                                 unsigned row_stride_elems,
                                                 unsigned tensor_w, unsigned tensor_h) {
    unsigned long long ga = (unsigned long long)(size_t)gaddr;
    u32x4 g0;
    g0[0] = 1u;                                        // count=1 (valid), user mode
    g0[1] = lds_addr;                                  // LDS byte address
    g0[2] = (unsigned)(ga & 0xFFFFFFFFu);              // global_addr[31:0]
    g0[3] = (unsigned)((ga >> 32) & 0x01FFFFFFu)       // global_addr[56:32]
          | (2u << 30);                                // type=2 ("image")
    i32x8 g1;
    g1[0] = (int)(1u << 16);                           // wg_mask=0, data_size=1 (2B)
    g1[1] = (int)((tensor_w & 0xFFFFu) << 16);         // tensor_dim0[15:0]
    g1[2] = (int)((tensor_w >> 16) | ((tensor_h & 0xFFFFu) << 16));  // dim0 hi | dim1 lo
    g1[3] = (int)((tensor_h >> 16) | ((tile_w & 0xFFFFu) << 16));    // dim1 hi | tile_dim0
    g1[4] = (int)(tile_h & 0xFFFFu);                   // tile_dim1 (tile_dim2=0)
    g1[5] = (int)row_stride_elems;                     // tensor_dim0_stride[31:0]
    g1[6] = 0;                                         // stride hi, dim1_stride lo
    g1[7] = 0;
    i32x4 z4 = {0, 0, 0, 0};
#if __clang_major__ >= 23
    i32x8 z8 = {0, 0, 0, 0, 0, 0, 0, 0};
    __builtin_amdgcn_tensor_load_to_lds(g0, g1, z4, z4, z8, 0);
#else
    __builtin_amdgcn_tensor_load_to_lds(g0, g1, z4, z4, 0);
#endif
}

// ---------------------------------------------------------------------------
// Weight convert + transpose: W[k][256] f32 (row-major)  ->  Wt[n][Kdim] bf16
// ---------------------------------------------------------------------------
__global__ void conv_wt_kernel(const float* __restrict__ W,
                               unsigned short* __restrict__ Wt, int Kdim) {
    int id = blockIdx.x * blockDim.x + threadIdx.x;
    if (id >= Kdim * 256) return;
    int k = id >> 8;
    int n = id & 255;
    Wt[(size_t)n * Kdim + k] = (unsigned short)f32_to_bf16(W[id]);
}

__global__ void fill_u32_kernel(unsigned int* __restrict__ p, unsigned int v, int n) {
    int id = blockIdx.x * blockDim.x + threadIdx.x;
    if (id < n) p[id] = v;
}

// ---------------------------------------------------------------------------
// msg[e] = [ h_n[src[e]] * h_e[e] , h_n[src[e]] ]  -> bf16 [E, 512]
// ---------------------------------------------------------------------------
__global__ void build_msg_kernel(const float* __restrict__ hn,
                                 const float* __restrict__ he,
                                 const int* __restrict__ src,
                                 unsigned short* __restrict__ msg, int E) {
    int id = blockIdx.x * blockDim.x + threadIdx.x;
    if (id >= E * 64) return;
    int e = id >> 6;
    int f = (id & 63) << 2;
    int s = src[e];
    float4 a = *(const float4*)(hn + (size_t)s * DDIM + f);
    float4 b = *(const float4*)(he + (size_t)e * DDIM + f);
    uint2 mp, mr;
    mp.x = f32_to_bf16(a.x * b.x) | (f32_to_bf16(a.y * b.y) << 16);
    mp.y = f32_to_bf16(a.z * b.z) | (f32_to_bf16(a.w * b.w) << 16);
    mr.x = f32_to_bf16(a.x)       | (f32_to_bf16(a.y) << 16);
    mr.y = f32_to_bf16(a.z)       | (f32_to_bf16(a.w) << 16);
    *(uint2*)(msg + (size_t)e * 512 + f)       = mp;
    *(uint2*)(msg + (size_t)e * 512 + 256 + f) = mr;
}

// ---------------------------------------------------------------------------
// Generic bf16 WMMA GEMM:  out[M,256] = buildA(mode)[M,Kdim] @ W + bias
//   MODE 0: A0 = f32 [M, Kdim]                                   (Q gemm)
//   MODE 1: A0 = bf16 [M, Kdim] prebuilt; M % 64 == 0; A via TDM (K/V gemm)
//   MODE 2: A = [A0 | A1], both f32 [M,256], Kdim=512            (Ww gemm)
//   MODE 3: A = [A0[idx0[r]] | A0[idx1[r]] | A1[r]], Kdim=768    (Ew gemm)
// B (weights) tile is always streamed by the Tensor Data Mover.
// LDS double-buffered: TDM for tile kt+1 overlaps WMMA compute of tile kt
// (TENSORcnt is in-order per wave, so wait tensorcnt<=OPS_per_tile retires kt).
// Block: 256 thr = 8 waves; block tile 64x256; wave tile 16x128.
// ---------------------------------------------------------------------------
template <int MODE>
__global__ __launch_bounds__(256)
void gemm_bf16_kernel(const void* __restrict__ A0v, const float* __restrict__ A1,
                      const int* __restrict__ idx0, const int* __restrict__ idx1,
                      const unsigned short* __restrict__ Wt,   // [256][Kdim] bf16
                      const float* __restrict__ bias,
                      float* __restrict__ out, int M, int Kdim) {
    __shared__ __align__(16) unsigned short Alds[2][64 * 32];   //  2 x  4 KB
    __shared__ __align__(16) unsigned short Blds[2][256 * 32];  //  2 x 16 KB

    constexpr int TOPS = (MODE == 1) ? 2 : 1;   // tensor ops per staged tile

    const int tid  = threadIdx.x;
    const int lane = tid & 31;
    const int w    = tid >> 5;
    const int wm   = w & 3;        // M tile within block (0..3)
    const int wn   = w >> 2;       // N half (0..1)
    const int row0 = blockIdx.x * 64;

    v8f acc[8] = {};

    // A-tile builder coords: thread -> (row 0..63, 8-wide k chunk)
    const int ar   = tid >> 2;
    const int ac0  = (tid & 3) * 8;
    const int grow = row0 + ar;

    // gather indices are K-loop invariant (MODE 3)
    int i0 = 0, i1 = 0;
    if (MODE == 3 && grow < M) {
        i0 = idx0[grow];
        i1 = idx1[grow];
    }

    // fragment coords (ISA 16-bit A 16x32 layout)
    const int frow = lane & 15;
    const int kb   = (lane >> 4) * 8;

    // stage tile kt into LDS buffer `buf`
    auto stage = [&](int kt, int buf) {
        const int k0 = kt << 5;
        if (tid < 32) {
            tdm_load_2d_bf16((unsigned)(size_t)(&Blds[buf][0]),
                             (const char*)Wt + (size_t)k0 * 2,
                             /*tile_w*/32, /*tile_h*/256,
                             /*stride*/(unsigned)Kdim,
                             /*tensor_w*/(unsigned)Kdim, /*tensor_h*/256);
            if (MODE == 1) {
                tdm_load_2d_bf16((unsigned)(size_t)(&Alds[buf][0]),
                                 (const char*)A0v + ((size_t)row0 * Kdim + k0) * 2,
                                 /*tile_w*/32, /*tile_h*/64,
                                 /*stride*/(unsigned)Kdim,
                                 /*tensor_w*/(unsigned)Kdim, /*tensor_h*/0x40000000u);
            }
        }
        if (MODE != 1) {
            const float* A0 = (const float*)A0v;
            uint4 pk = make_uint4(0u, 0u, 0u, 0u);
            if (grow < M) {
                const int f0 = k0 + ac0;          // 8-chunk never crosses a 256 seam
                const float* sp;
                if (MODE == 0) {
                    sp = A0 + (size_t)grow * Kdim + f0;
                } else if (MODE == 2) {
                    sp = (f0 < 256) ? (A0 + (size_t)grow * 256 + f0)
                                    : (A1 + (size_t)grow * 256 + (f0 - 256));
                } else { // MODE == 3
                    if (f0 < 256)
                        sp = A0 + (size_t)i0 * 256 + f0;
                    else if (f0 < 512)
                        sp = A0 + (size_t)i1 * 256 + (f0 - 256);
                    else
                        sp = A1 + (size_t)grow * 256 + (f0 - 512);
                }
                float4 p0 = ((const float4*)sp)[0];
                float4 p1 = ((const float4*)sp)[1];
                pk.x = f32_to_bf16(p0.x) | (f32_to_bf16(p0.y) << 16);
                pk.y = f32_to_bf16(p0.z) | (f32_to_bf16(p0.w) << 16);
                pk.z = f32_to_bf16(p1.x) | (f32_to_bf16(p1.y) << 16);
                pk.w = f32_to_bf16(p1.z) | (f32_to_bf16(p1.w) << 16);
            }
            *(uint4*)(&Alds[buf][ar * 32 + ac0]) = pk;
        }
    };

    const int nk = Kdim >> 5;
    stage(0, 0);                                   // prologue: tile 0 in flight
    for (int kt = 0; kt < nk; ++kt) {
        const int cur = kt & 1;
        if (kt + 1 < nk) {
            stage(kt + 1, cur ^ 1);                // overlap: next tile streams in
            if (tid < 32) __builtin_amdgcn_s_wait_tensorcnt(TOPS);  // tile kt retired
        } else {
            if (tid < 32) __builtin_amdgcn_s_wait_tensorcnt(0);
        }
        __syncthreads();                           // tile kt visible to all waves

        // --- compute: preload A frag + all 8 B frags, then 8 back-to-back WMMAs
        const unsigned short* Ab = &Alds[cur][0];
        const unsigned short* Bb = &Blds[cur][0];
        FragU af, bfr[8];
        af.u[0] = *(const uint4*)(Ab + (wm * 16 + frow) * 32 + kb);
        af.u[1] = *(const uint4*)(Ab + (wm * 16 + frow) * 32 + 16 + kb);
#pragma unroll
        for (int j = 0; j < 8; ++j) {
            const int n = (wn * 8 + j) * 16 + frow;
            bfr[j].u[0] = *(const uint4*)(Bb + n * 32 + kb);
            bfr[j].u[1] = *(const uint4*)(Bb + n * 32 + 16 + kb);
        }
#pragma unroll
        for (int j = 0; j < 8; ++j) {
            acc[j] = __builtin_amdgcn_wmma_f32_16x16x32_bf16(
                false, af.v, false, bfr[j].v, (short)0, acc[j], false, false);
        }
        __syncthreads();                           // release buffer `cur` for restage
    }

    // --- epilogue: C layout (VGPR r: lanes 0-15 -> M=r, lanes 16-31 -> M=8+r)
    const int col  = lane & 15;
    const int rsel = (lane >> 4) * 8;
#pragma unroll
    for (int j = 0; j < 8; ++j) {
        const int gcol = (wn * 8 + j) * 16 + col;
        const float bv = bias[gcol];
#pragma unroll
        for (int r = 0; r < 8; ++r) {
            const int orow = row0 + wm * 16 + rsel + r;
            if (orow < M)
                out[(size_t)orow * DDIM + gcol] = acc[j][r] + bv;
        }
    }
}

// ---------------------------------------------------------------------------
// alpha = Q[dst]*K (in-place into K), segment-max via encoded atomicMax
// ---------------------------------------------------------------------------
__global__ void alpha_max_kernel(const float* __restrict__ Q, float* __restrict__ K,
                                 const int* __restrict__ dst,
                                 unsigned int* __restrict__ menc, int E) {
    int id = blockIdx.x * blockDim.x + threadIdx.x;
    if (id >= E * 64) return;
    int e = id >> 6, f = (id & 63) << 2;
    int d = dst[e];
    float4 q = *(const float4*)(Q + (size_t)d * DDIM + f);
    float4 k = *(float4*)(K + (size_t)e * DDIM + f);
    float4 a = make_float4(q.x * k.x, q.y * k.y, q.z * k.z, q.w * k.w);
    *(float4*)(K + (size_t)e * DDIM + f) = a;
    unsigned int* mp = menc + (size_t)d * DDIM + f;
    atomicMax(mp + 0, enc_f32(a.x));
    atomicMax(mp + 1, enc_f32(a.y));
    atomicMax(mp + 2, enc_f32(a.z));
    atomicMax(mp + 3, enc_f32(a.w));
}

__global__ void exp_sum_kernel(float* __restrict__ K, const int* __restrict__ dst,
                               const unsigned int* __restrict__ menc,
                               float* __restrict__ z, int E) {
    int id = blockIdx.x * blockDim.x + threadIdx.x;
    if (id >= E * 64) return;
    int e = id >> 6, f = (id & 63) << 2;
    int d = dst[e];
    float4 a = *(float4*)(K + (size_t)e * DDIM + f);
    const unsigned int* mp = menc + (size_t)d * DDIM + f;
    float4 ex = make_float4(__expf(a.x - dec_f32(mp[0])), __expf(a.y - dec_f32(mp[1])),
                            __expf(a.z - dec_f32(mp[2])), __expf(a.w - dec_f32(mp[3])));
    *(float4*)(K + (size_t)e * DDIM + f) = ex;
    float* zp = z + (size_t)d * DDIM + f;
    atomicAdd(zp + 0, ex.x);
    atomicAdd(zp + 1, ex.y);
    atomicAdd(zp + 2, ex.z);
    atomicAdd(zp + 3, ex.w);
}

__global__ void agg_kernel(const float* __restrict__ K, const float* __restrict__ V,
                           const int* __restrict__ dst, const float* __restrict__ z,
                           float* __restrict__ agg, int E) {
    int id = blockIdx.x * blockDim.x + threadIdx.x;
    if (id >= E * 64) return;
    int e = id >> 6, f = (id & 63) << 2;
    int d = dst[e];
    float4 ex = *(const float4*)(K + (size_t)e * DDIM + f);
    float4 v  = *(const float4*)(V + (size_t)e * DDIM + f);
    const float* zp = z + (size_t)d * DDIM + f;
    float* ap = agg + (size_t)d * DDIM + f;
    atomicAdd(ap + 0, ex.x / zp[0] * v.x);
    atomicAdd(ap + 1, ex.y / zp[1] * v.y);
    atomicAdd(ap + 2, ex.z / zp[2] * v.z);
    atomicAdd(ap + 3, ex.w / zp[3] * v.w);
}

// ---------------------------------------------------------------------------
// relu -> layernorm, one wave32 per row (8 features / lane)
// ---------------------------------------------------------------------------
__global__ void relu_ln_kernel(const float* __restrict__ x, const float* __restrict__ g,
                               const float* __restrict__ b, float* __restrict__ out, int M) {
    int row  = blockIdx.x * 8 + (threadIdx.x >> 5);
    int lane = threadIdx.x & 31;
    if (row >= M) return;
    const float* xr = x + (size_t)row * DDIM + lane * 8;
    float4 p0 = *(const float4*)(xr);
    float4 p1 = *(const float4*)(xr + 4);
    float v[8] = {p0.x, p0.y, p0.z, p0.w, p1.x, p1.y, p1.z, p1.w};
    float s = 0.f, sq = 0.f;
#pragma unroll
    for (int i = 0; i < 8; ++i) {
        v[i] = fmaxf(v[i], 0.f);
        s += v[i];
        sq += v[i] * v[i];
    }
    for (int m = 16; m > 0; m >>= 1) {
        s  += __shfl_xor(s, m, 32);
        sq += __shfl_xor(sq, m, 32);
    }
    float mu  = s * (1.f / 256.f);
    float var = sq * (1.f / 256.f) - mu * mu;
    float rs  = rsqrtf(var + 1e-5f);
    float* op = out + (size_t)row * DDIM + lane * 8;
    const float* gp = g + lane * 8;
    const float* bp = b + lane * 8;
#pragma unroll
    for (int i = 0; i < 8; ++i)
        op[i] = (v[i] - mu) * rs * gp[i] + bp[i];
}

// ---------------------------------------------------------------------------
extern "C" void kernel_launch(void* const* d_in, const int* in_sizes, int n_in,
                              void* d_out, int out_size, void* d_ws, size_t ws_size,
                              hipStream_t stream) {
    const float* node_h = (const float*)d_in[0];
    const float* edge_h = (const float*)d_in[1];
    const int*   src    = (const int*)d_in[2];
    const int*   dst    = (const int*)d_in[3];
    const float* Kw = (const float*)d_in[4];
    const float* Kb = (const float*)d_in[5];
    const float* Vw = (const float*)d_in[6];
    const float* Vb = (const float*)d_in[7];
    const float* Qw = (const float*)d_in[8];
    const float* Qb = (const float*)d_in[9];
    const float* Ww = (const float*)d_in[10];
    const float* Wb = (const float*)d_in[11];
    const float* Ew = (const float*)d_in[12];
    const float* Eb = (const float*)d_in[13];
    const float* ln_g = (const float*)d_in[14];
    const float* ln_b = (const float*)d_in[15];

    const int N = N_NODES, E = N_EDGES;

    // workspace carve-up (256B aligned)
    char* p = (char*)d_ws;
    auto take = [&](size_t bytes) -> char* {
        char* r = p;
        p += (bytes + 255) & ~(size_t)255;
        return r;
    };
    unsigned short* wq  = (unsigned short*)take((size_t)NLAYERS * 256 * 256 * 2);
    unsigned short* wk  = (unsigned short*)take((size_t)NLAYERS * 512 * 256 * 2);
    unsigned short* wv  = (unsigned short*)take((size_t)NLAYERS * 512 * 256 * 2);
    unsigned short* ww  = (unsigned short*)take((size_t)NLAYERS * 512 * 256 * 2);
    unsigned short* we  = (unsigned short*)take((size_t)NLAYERS * 768 * 256 * 2);
    unsigned short* msg = (unsigned short*)take((size_t)E * 512 * 2);
    float* Qbuf = (float*)take((size_t)N * DDIM * 4);
    float* Kbuf = (float*)take((size_t)E * DDIM * 4);
    float* Vbuf = (float*)take((size_t)E * DDIM * 4);
    unsigned int* menc = (unsigned int*)take((size_t)N * DDIM * 4);
    float* zbuf = (float*)take((size_t)N * DDIM * 4);
    float* aggb = (float*)take((size_t)N * DDIM * 4);
    float* hnA  = (float*)take((size_t)N * DDIM * 4);
    float* hnB  = (float*)take((size_t)N * DDIM * 4);
    float* heA  = (float*)take((size_t)E * DDIM * 4);
    float* heB  = (float*)take((size_t)E * DDIM * 4);

    // convert + transpose weights to bf16 (both layers)
    for (int l = 0; l < NLAYERS; ++l) {
        conv_wt_kernel<<<(256 * 256 + 255) / 256, 256, 0, stream>>>(Qw + (size_t)l * 256 * 256, wq + (size_t)l * 256 * 256, 256);
        conv_wt_kernel<<<(512 * 256 + 255) / 256, 256, 0, stream>>>(Kw + (size_t)l * 512 * 256, wk + (size_t)l * 512 * 256, 512);
        conv_wt_kernel<<<(512 * 256 + 255) / 256, 256, 0, stream>>>(Vw + (size_t)l * 512 * 256, wv + (size_t)l * 512 * 256, 512);
        conv_wt_kernel<<<(512 * 256 + 255) / 256, 256, 0, stream>>>(Ww + (size_t)l * 512 * 256, ww + (size_t)l * 512 * 256, 512);
        conv_wt_kernel<<<(768 * 256 + 255) / 256, 256, 0, stream>>>(Ew + (size_t)l * 768 * 256, we + (size_t)l * 768 * 256, 768);
    }

    const int gN = (N + 63) / 64;     // 157
    const int gE = (E + 63) / 64;     // 2500
    const int gEF = (E * 64) / 256;   // 40000
    const int gNF = (N * DDIM + 255) / 256;

    const float* hn_in = node_h;
    const float* he_in = edge_h;
    for (int l = 0; l < NLAYERS; ++l) {
        // Q = h_n @ Qw + Qb
        gemm_bf16_kernel<0><<<gN, 256, 0, stream>>>(hn_in, nullptr, nullptr, nullptr,
            wq + (size_t)l * 256 * 256, Qb + l * DDIM, Qbuf, N, 256);
        // msg (bf16) = [src_h * h_e, src_h]
        build_msg_kernel<<<gEF, 256, 0, stream>>>(hn_in, he_in, src, msg, E);
        // K, V = msg @ {Kw,Vw}   (A and B tiles via TDM, double-buffered)
        gemm_bf16_kernel<1><<<gE, 256, 0, stream>>>(msg, nullptr, nullptr, nullptr,
            wk + (size_t)l * 512 * 256, Kb + l * DDIM, Kbuf, E, 512);
        gemm_bf16_kernel<1><<<gE, 256, 0, stream>>>(msg, nullptr, nullptr, nullptr,
            wv + (size_t)l * 512 * 256, Vb + l * DDIM, Vbuf, E, 512);
        // segment softmax state
        fill_u32_kernel<<<gNF, 256, 0, stream>>>(menc, 0x007FFFFFu, N * DDIM);  // enc(-inf)
        fill_u32_kernel<<<gNF, 256, 0, stream>>>((unsigned int*)zbuf, 0u, N * DDIM);
        fill_u32_kernel<<<gNF, 256, 0, stream>>>((unsigned int*)aggb, 0u, N * DDIM);
        alpha_max_kernel<<<gEF, 256, 0, stream>>>(Qbuf, Kbuf, dst, menc, E);
        exp_sum_kernel<<<gEF, 256, 0, stream>>>(Kbuf, dst, menc, zbuf, E);
        agg_kernel<<<gEF, 256, 0, stream>>>(Kbuf, Vbuf, dst, zbuf, aggb, E);
        // h_n_new (raw) = [agg, h_n] @ Ww + Wb
        gemm_bf16_kernel<2><<<gN, 256, 0, stream>>>(aggb, hn_in, nullptr, nullptr,
            ww + (size_t)l * 512 * 256, Wb + l * DDIM, hnB, N, 512);
        // h_e_new (raw) = [h_nn[src], h_nn[dst], h_e] @ Ew + Eb  (gather fused in A build)
        gemm_bf16_kernel<3><<<gE, 256, 0, stream>>>(hnB, he_in, src, dst,
            we + (size_t)l * 768 * 256, Eb + l * DDIM, heB, E, 768);
        // relu + layernorm
        float* hn_out = (l == NLAYERS - 1) ? (float*)d_out : hnA;
        float* he_out = (l == NLAYERS - 1) ? ((float*)d_out + (size_t)N * DDIM) : heA;
        relu_ln_kernel<<<(N + 7) / 8, 256, 0, stream>>>(hnB, ln_g, ln_b, hn_out, N);
        relu_ln_kernel<<<(E + 7) / 8, 256, 0, stream>>>(heB, ln_g, ln_b, he_out, E);
        hn_in = hn_out;
        he_in = he_out;
    }
}
